// UnimodalAttentionKernel_88562225643877
// MI455X (gfx1250) — matrix operand order repaired and checked by reading, not compiled, not run on hardware
//
#include <hip/hip_runtime.h>
#include <hip/hip_bf16.h>

// B=8, S=1024, D=1024, H=16, HD=64
// Pipeline:
//   1) q = (query @ Wq^T + bq) * 0.125   -> f16 head-major [b][h][s][hd]   (WMMA GEMM)
//   2) k = key   @ Wk^T + bk             -> f16 head-major                 (WMMA GEMM)
//   3) v = value @ Wv^T + bv             -> f16 head-major                 (WMMA GEMM)
//   4) flash attention (causal, online softmax) -> ctx f16 [b][s][h][hd]   (WMMA)
//   5) out = ctx @ Wo^T + bo             -> f32 [b*s][d]                   (WMMA GEMM,
//      A-tile staged to LDS via global_load_async_to_lds_b128 inline asm)

typedef __attribute__((ext_vector_type(16))) _Float16 v16h;
typedef __attribute__((ext_vector_type(8)))  float    v8f;
typedef __attribute__((ext_vector_type(4)))  unsigned int u32x4;

#define WMMA_F16(a, b, c) \
  __builtin_amdgcn_wmma_f32_16x16x32_f16(false, (a), false, (b), (short)0, (c), false, false)

__device__ __forceinline__ v16h load_frag_pair(const _Float16* p0, const _Float16* p1) {
  union { v16h h; u32x4 q[2]; } u;
  u.q[0] = *reinterpret_cast<const u32x4*>(p0);
  u.q[1] = *reinterpret_cast<const u32x4*>(p1);
  return u.h;
}

// gfx1250 async global->LDS 16B copy (ASYNCcnt-tracked).
// lds_addr = low 32 bits of generic shared pointer (ISA 10.2: LDS aperture).
__device__ __forceinline__ void async_copy_b128(const void* gp, void* lp) {
  unsigned lds_addr = (unsigned)(uintptr_t)lp;
  asm volatile("global_load_async_to_lds_b128 %0, %1, off"
               :
               : "v"(lds_addr), "v"(gp)
               : "memory");
}
__device__ __forceinline__ void wait_asynccnt0() {
  asm volatile("s_wait_asynccnt 0" ::: "memory");
}

// ---------------------------------------------------------------------------
// GEMM: C[M x 1024] = A[M x 1024] @ W^T + bias, W is [1024 x 1024] row-major.
// A_F16:  A is f16 (ctx), else f32 (raw inputs, converted on stage).
// OUT_F32: plain f32 row-major store; else f16 head-major store with scale.
// Workgroup = 256 thr (8 waves), tile 128x128, K-chunk 32.
// ---------------------------------------------------------------------------
template <bool A_F16, bool OUT_F32>
__global__ __launch_bounds__(256) void cdna5_gemm_kernel(
    const void* __restrict__ Aall, const float* __restrict__ W,
    const float* __restrict__ bias, void* __restrict__ Out, float scale) {
  constexpr int K = 1024;
  constexpr int LD = 40;  // LDS row stride in halves (pad vs bank conflicts)
  __shared__ __attribute__((aligned(16))) _Float16 As[128 * LD];
  __shared__ __attribute__((aligned(16))) _Float16 Bs[128 * LD];

  const int tid  = threadIdx.x;
  const int wave = tid >> 5, lane = tid & 31;
  const int wy = wave >> 1, wx = wave & 1;        // 4x2 wave grid
  const int half = lane >> 4, l16 = lane & 15;
  const int m0 = blockIdx.y * 128;
  const int n0 = blockIdx.x * 128;
  const int a_kb = 8 * half;    // A-frag k base within 32-chunk
  const int b_kk = 8 * half;    // B-frag k base within 32-chunk

  v8f acc[2][4];
  for (int mi = 0; mi < 2; ++mi)
    for (int ni = 0; ni < 4; ++ni) acc[mi][ni] = v8f{};

  for (int k0 = 0; k0 < K; k0 += 32) {
    // ---- stage W tile (128 rows of W = B columns), f32 -> f16 ----
    for (int it = 0; it < 4; ++it) {
      int idx = it * 1024 + tid * 4;          // 4096 elems total
      int r = idx >> 5, c = idx & 31;
      float4 wv = *reinterpret_cast<const float4*>(
          W + (size_t)(n0 + r) * 1024 + k0 + c);
      _Float16* d = &Bs[r * LD + c];
      d[0] = (_Float16)wv.x; d[1] = (_Float16)wv.y;
      d[2] = (_Float16)wv.z; d[3] = (_Float16)wv.w;
    }
    // ---- stage A tile (128 x 32) into LDS as f16 ----
    if constexpr (A_F16) {
      // gfx1250 async DMA path: 512 16-byte chunks, 2 per thread.
      for (int it = 0; it < 2; ++it) {
        int c = it * 256 + tid;
        int r = c >> 2, col = (c & 3) * 8;
        const _Float16* gp =
            (const _Float16*)Aall + (size_t)(m0 + r) * 1024 + k0 + col;
        async_copy_b128(gp, &As[r * LD + col]);
      }
      wait_asynccnt0();
    } else {
      for (int it = 0; it < 4; ++it) {
        int idx = it * 1024 + tid * 4;
        int r = idx >> 5, c = idx & 31;
        float4 v = *reinterpret_cast<const float4*>(
            (const float*)Aall + (size_t)(m0 + r) * 1024 + k0 + c);
        _Float16* d = &As[r * LD + c];
        d[0] = (_Float16)v.x; d[1] = (_Float16)v.y;
        d[2] = (_Float16)v.z; d[3] = (_Float16)v.w;
      }
    }
    __syncthreads();

    // prefetch next K tile (gfx1250 global_prefetch_b8)
    if (k0 + 32 < K) {
      const size_t esz = A_F16 ? 2 : 4;
      __builtin_prefetch((const char*)Aall +
                             ((size_t)(m0 + (tid >> 1)) * 1024 + k0 + 32) * esz,
                         0, 1);
      __builtin_prefetch(
          (const char*)(W + (size_t)(n0 + (tid >> 1)) * 1024 + k0 + 32), 0, 1);
    }

    // ---- fragments + WMMA ----
    v16h af[2], bf[4];
    for (int mi = 0; mi < 2; ++mi) {
      const _Float16* p = &As[(wy * 32 + mi * 16 + l16) * LD + a_kb];
      af[mi] = load_frag_pair(p, p + 16);
    }
    for (int ni = 0; ni < 4; ++ni) {
      const _Float16* p = &Bs[(wx * 64 + ni * 16 + l16) * LD + b_kk];
      bf[ni] = load_frag_pair(p, p + 16);
    }
    for (int mi = 0; mi < 2; ++mi)
      for (int ni = 0; ni < 4; ++ni)
        acc[mi][ni] = WMMA_F16(af[mi], bf[ni], acc[mi][ni]);
    __syncthreads();
  }

  // ---- epilogue ----
  for (int mi = 0; mi < 2; ++mi) {
    const int rbase = m0 + wy * 32 + mi * 16 + 8 * half;
    for (int ni = 0; ni < 4; ++ni) {
      const int col = n0 + wx * 64 + ni * 16 + l16;
      const float bv = bias[col];
      for (int r = 0; r < 8; ++r) {
        float v = acc[mi][ni][r] + bv;
        const int row = rbase + r;
        if constexpr (OUT_F32) {
          ((float*)Out)[(size_t)row * 1024 + col] = v;
        } else {
          v *= scale;
          const int b = row >> 10, s = row & 1023;
          const int h = col >> 6, hd = col & 63;
          ((_Float16*)Out)[(((size_t)(b * 16 + h)) * 1024 + s) * 64 + hd] =
              (_Float16)v;
        }
      }
    }
  }
}

// ---------------------------------------------------------------------------
// Flash attention (causal). Q/K/V f16 head-major [bh][s][64]. 128 thr = 4
// waves; each wave owns 16 query rows; WG covers 64 rows. Keys streamed in
// 32-wide blocks; V transposed into LDS for PV B-fragments; P bounced through
// per-wave LDS to convert C-layout -> A-fragment layout (wave-synchronous DS).
// ---------------------------------------------------------------------------
__global__ __launch_bounds__(128) void cdna5_flash_attn_kernel(
    const _Float16* __restrict__ Q, const _Float16* __restrict__ Kh,
    const _Float16* __restrict__ V, _Float16* __restrict__ Ctx) {
  constexpr int S = 1024, HD = 64, LD = 40;
  __shared__ __attribute__((aligned(16))) _Float16 Vt[64 * LD];      // [hd][key]
  __shared__ __attribute__((aligned(16))) _Float16 Pbuf[4][16 * LD]; // per wave

  const int tid = threadIdx.x, wave = tid >> 5, lane = tid & 31;
  const int half = lane >> 4, l16 = lane & 15;
  const int bh = blockIdx.y;                  // b*16 + h
  const int b = bh >> 4, h = bh & 15;
  const int qbase = blockIdx.x * 64;
  const int qr0 = qbase + wave * 16;
  const _Float16* qh = Q + (size_t)bh * S * HD;
  const _Float16* kh = Kh + (size_t)bh * S * HD;
  const _Float16* vh = V + (size_t)bh * S * HD;

  // Q A-fragments (16 rows x 64, two k-chunks of 32)
  v16h aq[2];
  {
    const _Float16* p = qh + (size_t)(qr0 + l16) * HD;
    for (int g = 0; g < 2; ++g) {
      const _Float16* pg = p + g * 32 + 8 * half;
      aq[g] = load_frag_pair(pg, pg + 16);
    }
  }

  v8f o[4];
  for (int n = 0; n < 4; ++n) o[n] = v8f{};
  float mrow[8], lrow[8];
  for (int r = 0; r < 8; ++r) { mrow[r] = -1e30f; lrow[r] = 0.f; }

  const int nkb = (qbase + 64) >> 5;  // causal: keys <= qbase+63
  for (int kb = 0; kb < nkb; ++kb) {
    const int k0 = kb * 32;
    // ---- cooperative transpose stage of V block (32 keys x 64 hd) ----
    for (int it = 0; it < 2; ++it) {
      int c = it * 128 + tid;      // 256 chunks of 8 halves
      int r = c >> 3;              // key row 0..31
      int cg = (c & 7) * 8;        // hd base
      union { u32x4 q; _Float16 hv[8]; } u;
      u.q = *reinterpret_cast<const u32x4*>(vh + (size_t)(k0 + r) * HD + cg);
      for (int j = 0; j < 8; ++j) Vt[(cg + j) * LD + r] = u.hv[j];
    }
    __syncthreads();

    if (k0 <= qr0 + 15) {
      // ---- scores: two 16x16 tiles over 32 keys ----
      v8f sc[2];
      for (int c = 0; c < 2; ++c) {
        v8f z = v8f{};
        for (int g = 0; g < 2; ++g) {
          const _Float16* p =
              kh + (size_t)(k0 + c * 16 + l16) * HD + g * 32 + 8 * half;
          v16h bk = load_frag_pair(p, p + 16);
          z = WMMA_F16(aq[g], bk, z);
        }
        sc[c] = z;
      }
      // ---- causal mask ----
      for (int c = 0; c < 2; ++c) {
        const int key = k0 + c * 16 + l16;
        for (int r = 0; r < 8; ++r)
          if (key > qr0 + r + 8 * half) sc[c][r] = -1e30f;
      }
      // ---- online softmax (reduce across 16-lane halves) ----
      float alpha[8];
      for (int r = 0; r < 8; ++r) {
        float mx = fmaxf(sc[0][r], sc[1][r]);
        for (int off = 1; off < 16; off <<= 1)
          mx = fmaxf(mx, __shfl_xor(mx, off, 32));
        const float mnew = fmaxf(mrow[r], mx);
        alpha[r] = __expf(mrow[r] - mnew);
        mrow[r] = mnew;
        float rs = 0.f;
        for (int c = 0; c < 2; ++c) {
          float p = __expf(sc[c][r] - mnew);
          sc[c][r] = p;
          rs += p;
        }
        for (int off = 1; off < 16; off <<= 1) rs += __shfl_xor(rs, off, 32);
        lrow[r] = lrow[r] * alpha[r] + rs;
      }
      // ---- P -> per-wave LDS -> A-fragment (wave-synchronous DS ordering) --
      _Float16* Pw = Pbuf[wave];
      for (int c = 0; c < 2; ++c)
        for (int r = 0; r < 8; ++r)
          Pw[(r + 8 * half) * LD + c * 16 + l16] = (_Float16)sc[c][r];
#if __has_builtin(__builtin_amdgcn_wave_barrier)
      __builtin_amdgcn_wave_barrier();
#endif
      v16h ap;
      {
        const _Float16* p = &Pw[l16 * LD + 8 * half];
        ap = load_frag_pair(p, p + 16);
      }
      // ---- rescale O, accumulate P @ V ----
      for (int n = 0; n < 4; ++n)
        for (int r = 0; r < 8; ++r) o[n][r] *= alpha[r];
      for (int n = 0; n < 4; ++n) {
        const _Float16* p = &Vt[(n * 16 + l16) * LD + 8 * half];
        v16h bv = load_frag_pair(p, p + 16);
        o[n] = WMMA_F16(ap, bv, o[n]);
      }
    }
    __syncthreads();
  }

  // ---- normalize and store ctx [b][s][h][hd] (GEMM-ready row-major) ----
  for (int n = 0; n < 4; ++n) {
    const int hd = n * 16 + l16;
    for (int r = 0; r < 8; ++r) {
      const int srow = qr0 + r + 8 * half;
      Ctx[(((size_t)(b * S + srow)) * 16 + h) * 64 + hd] =
          (_Float16)(o[n][r] / lrow[r]);
    }
  }
}

extern "C" void kernel_launch(void* const* d_in, const int* in_sizes, int n_in,
                              void* d_out, int out_size, void* d_ws,
                              size_t ws_size, hipStream_t stream) {
  const float* query = (const float*)d_in[0];
  const float* key   = (const float*)d_in[1];
  const float* value = (const float*)d_in[2];
  const float* Wq = (const float*)d_in[3];
  const float* bq = (const float*)d_in[4];
  const float* Wk = (const float*)d_in[5];
  const float* bk = (const float*)d_in[6];
  const float* Wv = (const float*)d_in[7];
  const float* bv = (const float*)d_in[8];
  const float* Wo = (const float*)d_in[9];
  const float* bo = (const float*)d_in[10];
  float* out = (float*)d_out;

  const size_t elems = (size_t)8 * 16 * 1024 * 64;  // B*H*S*HD
  _Float16* qws = (_Float16*)d_ws;
  _Float16* kws = qws + elems;
  _Float16* vws = kws + elems;
  _Float16* cws = vws + elems;

  dim3 gG(8, 64);  // (N/128, M/128), M = B*S = 8192
  cdna5_gemm_kernel<false, false><<<gG, 256, 0, stream>>>(
      (const void*)query, Wq, bq, (void*)qws, 0.125f);  // 1/sqrt(64)
  cdna5_gemm_kernel<false, false><<<gG, 256, 0, stream>>>(
      (const void*)key, Wk, bk, (void*)kws, 1.0f);
  cdna5_gemm_kernel<false, false><<<gG, 256, 0, stream>>>(
      (const void*)value, Wv, bv, (void*)vws, 1.0f);

  cdna5_flash_attn_kernel<<<dim3(16, 128), 128, 0, stream>>>(qws, kws, vws, cws);

  cdna5_gemm_kernel<true, true><<<gG, 256, 0, stream>>>(
      (const void*)cws, Wo, bo, (void*)out, 1.0f);
}
